// GeoTransformerLayer_46643344835053
// MI455X (gfx1250) — matrix operand order repaired and checked by reading, not compiled, not run on hardware
//
#include <hip/hip_runtime.h>
#include <hip/hip_bf16.h>
#include <stdint.h>

#define HIDDEN 128
#define HEADS 8

typedef __attribute__((ext_vector_type(16))) __bf16 v16bf;
typedef __attribute__((ext_vector_type(8)))  float  v8f;

union Frag16 { uint32_t u[8]; v16bf v; };

__device__ __forceinline__ uint32_t pack_bf16(float lo, float hi) {
  uint32_t ul = __float_as_uint(lo);
  uint32_t uh = __float_as_uint(hi);
  ul = (ul + 0x7fffu + ((ul >> 16) & 1u)) >> 16;   // round-to-nearest-even
  uh = (uh + 0x7fffu + ((uh >> 16) & 1u)) >> 16;
  return (uh << 16) | (ul & 0xffffu);
}

// ---------------------------------------------------------------------------
// Pack a weight matrix W (Nw x K, row-major, used as x @ W^T) into a contiguous-k (k = 16h + s)
// WMMA bf16 B-fragment layout: frag[((kt*NTtot + ntOff + ntl)*32 + lane)*8 + v]
// B element (k,n): lane = n%16 + 16*(k1/16), vgpr = (k1%16)/2, half = k1%2.
// ---------------------------------------------------------------------------
__global__ void pack_b(const float* __restrict__ W, uint32_t* __restrict__ frag,
                       int K, int Nw, int NTtot, int ntOff) {
  int KT = K / 32, NTloc = Nw / 16;
  int total = KT * NTloc * 256;
  int idx = blockIdx.x * blockDim.x + threadIdx.x;
  if (idx >= total) return;
  int v    = idx & 7;
  int lane = (idx >> 3) & 31;
  int ntl  = (idx >> 8) % NTloc;
  int kt   = (idx >> 8) / NTloc;
  int n = ntl * 16 + (lane & 15);
  int k = kt * 32 + ((lane >> 4) << 4) + (v << 1);
  float lo = W[(size_t)n * K + k];
  float hi = W[(size_t)n * K + k + 1];
  frag[(((size_t)kt * NTtot + ntOff + ntl) * 32 + lane) * 8 + v] = pack_bf16(lo, hi);
}

__global__ void concat3(const float* __restrict__ a, const float* __restrict__ b,
                        const float* __restrict__ c, float* __restrict__ o, int n) {
  int t = blockIdx.x * blockDim.x + threadIdx.x;
  if (t < n) o[t] = a[t];
  else if (t < 2 * n) o[t] = b[t - n];
  else if (t < 3 * n) o[t] = c[t - 2 * n];
}

__global__ void init_buffers(float* __restrict__ mbuf, float* __restrict__ denom,
                             float* __restrict__ agg, int nNodes) {
  int t = blockIdx.x * blockDim.x + threadIdx.x;
  if (t < nNodes * HEADS) { mbuf[t] = -3.402823466e38f; denom[t] = 0.f; }
  if (t < nNodes * HIDDEN) agg[t] = 0.f;
}

// ---------------------------------------------------------------------------
// C[M x N] = act( A[M x K](f32) * B(pre-packed bf16 frags) + bias )
// Block = 256 threads = 8 waves, each wave owns a 16-row tile (128 rows/blk).
// A is staged through LDS in 128x128 bf16 chunks; K is looped in 128-chunks.
// ---------------------------------------------------------------------------
template <int K, int N, bool GELU>
__global__ __launch_bounds__(256)
void gemm_bf16_wmma(const float* __restrict__ A, const uint32_t* __restrict__ Bfrag,
                    const float* __restrict__ bias, float* __restrict__ Cout, int M) {
  constexpr int ROWS = 128;
  constexpr int NT   = N / 16;
  constexpr int KCH  = K / 128;
  __shared__ uint32_t lds[ROWS * 64];   // 128 rows x 128 bf16 (pairs) = 32 KB
  const int wave = threadIdx.x >> 5;
  const int lane = threadIdx.x & 31;
  const int mblk = blockIdx.x * ROWS;

  auto loadChunk = [&](int kc) {
    for (int c8 = threadIdx.x; c8 < ROWS * 16; c8 += 256) {
      int row  = c8 >> 4;
      int col  = (c8 & 15) << 3;
      int grow = mblk + row;
      float4 f0 = {0.f, 0.f, 0.f, 0.f}, f1 = {0.f, 0.f, 0.f, 0.f};
      if (grow < M) {
        const float4* p = (const float4*)(A + (size_t)grow * K + (size_t)kc * 128 + col);
        f0 = p[0];
        f1 = p[1];
        if (KCH > 1 && kc + 1 < KCH)
          __builtin_prefetch(A + (size_t)grow * K + (size_t)(kc + 1) * 128 + col, 0, 1);
      }
      uint32_t* dst = lds + row * 64 + (col >> 1);
      dst[0] = pack_bf16(f0.x, f0.y);
      dst[1] = pack_bf16(f0.z, f0.w);
      dst[2] = pack_bf16(f1.x, f1.y);
      dst[3] = pack_bf16(f1.z, f1.w);
    }
  };

  // A fragment (16x32 bf16): lane group g: VGPR0-7 hold K=g*16..+15 pairs
  auto loadA = [&](Frag16& a, int kt4) {
    const uint32_t* Ar = lds + (wave * 16 + (lane & 15)) * 64;
    const int g8 = (lane >> 4) << 3;
#pragma unroll
    for (int i = 0; i < 8; ++i) {
      int k0 = kt4 * 32 + (g8 << 1) + (i << 1);
      a.u[i] = Ar[k0 >> 1];
    }
  };

  auto loadB = [&](Frag16& b, int ktg, int nt) {
    const uint4* p = (const uint4*)(Bfrag + (((size_t)ktg * NT + nt) * 32 + lane) * 8);
    *(uint4*)&b.u[0] = p[0];
    *(uint4*)&b.u[4] = p[1];
  };

  // Tile fully in-bounds? -> wave-uniform: single scalar branch, unguarded
  // stores on the fast path (all blocks except the one straddling M).
  const bool tileFull = (mblk + wave * 16 + 16) <= M;

  // C/D layout: VGPR r, lanes 0-15 -> M=r, lanes 16-31 -> M=8+r; N = lane%16.
  auto epilogue = [&](int nt, v8f c) {
    int n = nt * 16 + (lane & 15);
    float bb = bias[n];
    int mbase = mblk + wave * 16 + ((lane >> 4) << 3);
    float* cp = Cout + (size_t)mbase * N + n;
    if (tileFull) {
#pragma unroll
      for (int r = 0; r < 8; ++r) {
        float vv = c[r] + bb;
        if (GELU) vv = 0.5f * vv * (1.f + erff(vv * 0.70710678118654752f));
        cp[(size_t)r * N] = vv;
      }
    } else {
#pragma unroll
      for (int r = 0; r < 8; ++r) {
        if (mbase + r < M) {
          float vv = c[r] + bb;
          if (GELU) vv = 0.5f * vv * (1.f + erff(vv * 0.70710678118654752f));
          cp[(size_t)r * N] = vv;
        }
      }
    }
  };

  if constexpr (KCH == 1) {
    loadChunk(0);
    __syncthreads();
    for (int nt = 0; nt < NT; ++nt) {
      v8f c = {};
#pragma unroll
      for (int kt = 0; kt < 4; ++kt) {
        Frag16 a, b;
        loadA(a, kt);
        loadB(b, kt, nt);
        c = __builtin_amdgcn_wmma_f32_16x16x32_bf16(false, a.v, false, b.v,
                                                    (short)0, c, false, false);
      }
      epilogue(nt, c);
    }
  } else {
    v8f c[NT];
#pragma unroll
    for (int nt = 0; nt < NT; ++nt)
#pragma unroll
      for (int i = 0; i < 8; ++i) c[nt][i] = 0.f;
    for (int kc = 0; kc < KCH; ++kc) {
      if (kc) __syncthreads();
      loadChunk(kc);
      __syncthreads();
#pragma unroll
      for (int kt = 0; kt < 4; ++kt) {
        Frag16 a;
        loadA(a, kt);
#pragma unroll
        for (int nt = 0; nt < NT; ++nt) {
          Frag16 b;
          loadB(b, kc * 4 + kt, nt);
          c[nt] = __builtin_amdgcn_wmma_f32_16x16x32_bf16(false, a.v, false, b.v,
                                                          (short)0, c[nt], false, false);
        }
      }
    }
    for (int nt = 0; nt < NT; ++nt) epilogue(nt, c[nt]);
  }
}

// ---------------------------------------------------------------------------
// Edge phase: one thread per (edge, head). qkv = [N][384] (q|k|v).
// ---------------------------------------------------------------------------
__device__ __forceinline__ void atomicMaxF(float* addr, float val) {
  unsigned int* ua = (unsigned int*)addr;
  unsigned int old = *ua;
  while (__uint_as_float(old) < val) {
    unsigned int assumed = old;
    old = atomicCAS(ua, assumed, __float_as_uint(val));
    if (old == assumed) break;
  }
}

__global__ void edge_alpha(const float* __restrict__ qkv, const int* __restrict__ ei,
                           float* __restrict__ alpha, float* __restrict__ mbuf, int E) {
  int t = blockIdx.x * blockDim.x + threadIdx.x;
  if (t >= E * HEADS) return;
  int e = t >> 3, h = t & 7;
  int s = ei[e], d = ei[E + e];
  const float4* qp = (const float4*)(qkv + (size_t)d * 384 + h * 16);
  const float4* kp = (const float4*)(qkv + (size_t)s * 384 + 128 + h * 16);
  float dot = 0.f;
#pragma unroll
  for (int i = 0; i < 4; ++i) {
    float4 q = qp[i], k = kp[i];
    dot += q.x * k.x + q.y * k.y + q.z * k.z + q.w * k.w;
  }
  float av = dot * 0.25f;  // 1/sqrt(16)
  alpha[t] = av;
  atomicMaxF(mbuf + (size_t)d * HEADS + h, av);
}

__global__ void edge_exp(const int* __restrict__ ei, float* __restrict__ alpha,
                         const float* __restrict__ mbuf, float* __restrict__ denom, int E) {
  int t = blockIdx.x * blockDim.x + threadIdx.x;
  if (t >= E * HEADS) return;
  int e = t >> 3, h = t & 7;
  int d = ei[E + e];
  float ea = expf(alpha[t] - mbuf[(size_t)d * HEADS + h]);
  alpha[t] = ea;
  atomicAdd(denom + (size_t)d * HEADS + h, ea);
}

__global__ void edge_agg(const float* __restrict__ qkv, const int* __restrict__ ei,
                         const float* __restrict__ alpha, const float* __restrict__ denom,
                         float* __restrict__ agg, int E) {
  int t = blockIdx.x * blockDim.x + threadIdx.x;
  if (t >= E * HEADS) return;
  int e = t >> 3, h = t & 7;
  int s = ei[e], d = ei[E + e];
  float w = alpha[t] / (denom[(size_t)d * HEADS + h] + 1e-16f);
  const float4* vp = (const float4*)(qkv + (size_t)s * 384 + 256 + h * 16);
  float* ag = agg + (size_t)d * HIDDEN + h * 16;
#pragma unroll
  for (int i = 0; i < 4; ++i) {
    float4 vv = vp[i];
    atomicAdd(ag + 4 * i + 0, vv.x * w);
    atomicAdd(ag + 4 * i + 1, vv.y * w);
    atomicAdd(ag + 4 * i + 2, vv.z * w);
    atomicAdd(ag + 4 * i + 3, vv.w * w);
  }
}

// ---------------------------------------------------------------------------
// out[row] = LayerNorm(A[row] + B[row]) * g + b ; one wave32 per 128-wide row
// ---------------------------------------------------------------------------
__global__ __launch_bounds__(256)
void add_ln(const float* __restrict__ A, const float* __restrict__ B,
            const float* __restrict__ g, const float* __restrict__ b,
            float* __restrict__ out, int M) {
  int wave = threadIdx.x >> 5, lane = threadIdx.x & 31;
  int row = blockIdx.x * 8 + wave;
  if (row >= M) return;
  const float4* ap = (const float4*)(A + (size_t)row * HIDDEN);
  const float4* bp = (const float4*)(B + (size_t)row * HIDDEN);
  float4 va = ap[lane], vb = bp[lane];
  float v0 = va.x + vb.x, v1 = va.y + vb.y, v2 = va.z + vb.z, v3 = va.w + vb.w;
  float s = v0 + v1 + v2 + v3;
#pragma unroll
  for (int i = 16; i > 0; i >>= 1) s += __shfl_xor(s, i, 32);
  float mu = s * (1.f / 128.f);
  float d0 = v0 - mu, d1 = v1 - mu, d2 = v2 - mu, d3 = v3 - mu;
  float sq = d0 * d0 + d1 * d1 + d2 * d2 + d3 * d3;
#pragma unroll
  for (int i = 16; i > 0; i >>= 1) sq += __shfl_xor(sq, i, 32);
  float rs = rsqrtf(sq * (1.f / 128.f) + 1e-5f);
  int c = lane * 4;
  float* o = out + (size_t)row * HIDDEN + c;
  o[0] = d0 * rs * g[c + 0] + b[c + 0];
  o[1] = d1 * rs * g[c + 1] + b[c + 1];
  o[2] = d2 * rs * g[c + 2] + b[c + 2];
  o[3] = d3 * rs * g[c + 3] + b[c + 3];
}

// ---------------------------------------------------------------------------
extern "C" void kernel_launch(void* const* d_in, const int* in_sizes, int n_in,
                              void* d_out, int out_size, void* d_ws, size_t ws_size,
                              hipStream_t stream) {
  const float* x   = (const float*)d_in[0];
  const int*   ei  = (const int*)d_in[1];
  const float* Wq  = (const float*)d_in[2];
  const float* bq  = (const float*)d_in[3];
  const float* Wk  = (const float*)d_in[4];
  const float* bk  = (const float*)d_in[5];
  const float* Wv  = (const float*)d_in[6];
  const float* bv  = (const float*)d_in[7];
  const float* Wo  = (const float*)d_in[8];
  const float* bo  = (const float*)d_in[9];
  const float* W1  = (const float*)d_in[10];
  const float* b1  = (const float*)d_in[11];
  const float* W2  = (const float*)d_in[12];
  const float* b2  = (const float*)d_in[13];
  const float* g1  = (const float*)d_in[14];
  const float* be1 = (const float*)d_in[15];
  const float* g2  = (const float*)d_in[16];
  const float* be2 = (const float*)d_in[17];
  const int NN = in_sizes[0] / HIDDEN;   // 50000
  const int E  = in_sizes[1] / 2;        // 800000

  // ---- workspace layout ----
  char* base = (char*)d_ws;
  size_t off = 0;
  auto alloc = [&](size_t bytes) -> void* {
    void* p = base + off;
    off = (off + bytes + 255) & ~(size_t)255;
    return p;
  };
  uint32_t* wqkvF = (uint32_t*)alloc((size_t)4 * 24 * 256 * 4);   // K=128, NT=24 (q|k|v)
  uint32_t* woF   = (uint32_t*)alloc((size_t)4 * 8  * 256 * 4);   // K=128, NT=8
  uint32_t* w1F   = (uint32_t*)alloc((size_t)4 * 32 * 256 * 4);   // K=128, NT=32
  uint32_t* w2F   = (uint32_t*)alloc((size_t)16 * 8 * 256 * 4);   // K=512, NT=8
  float* bqkv  = (float*)alloc(384 * 4);
  float* mbuf  = (float*)alloc((size_t)NN * HEADS * 4);
  float* denom = (float*)alloc((size_t)NN * HEADS * 4);
  // big region: [qkv | alpha] during attention, reused as ff-hidden (tmid) after
  size_t qkvPad   = (((size_t)NN * 384 * 4) + 255) & ~(size_t)255;
  size_t bigBytes = qkvPad + (size_t)E * HEADS * 4;
  size_t tmid_b   = (size_t)NN * 512 * 4;
  if (tmid_b > bigBytes) bigBytes = tmid_b;
  char*  big   = (char*)alloc(bigBytes);
  float* qkv   = (float*)big;
  float* alpha = (float*)(big + qkvPad);
  float* tmid  = (float*)big;
  float* agg   = (float*)alloc((size_t)NN * HIDDEN * 4);
  float* att   = (float*)alloc((size_t)NN * HIDDEN * 4);   // also reused as ff2 out
  float* hbuf  = (float*)alloc((size_t)NN * HIDDEN * 4);

  // ---- weight packing (tiny, once per launch) ----
  auto packLaunch = [&](const float* W, uint32_t* frag, int K, int Nw, int NTtot, int ntOff) {
    int total = (K / 32) * (Nw / 16) * 256;
    pack_b<<<(total + 255) / 256, 256, 0, stream>>>(W, frag, K, Nw, NTtot, ntOff);
  };
  packLaunch(Wq, wqkvF, 128, 128, 24, 0);
  packLaunch(Wk, wqkvF, 128, 128, 24, 8);
  packLaunch(Wv, wqkvF, 128, 128, 24, 16);
  packLaunch(Wo, woF,   128, 128, 8, 0);
  packLaunch(W1, w1F,   128, 512, 32, 0);
  packLaunch(W2, w2F,   512, 128, 8, 0);
  concat3<<<2, 256, 0, stream>>>(bq, bk, bv, bqkv, HIDDEN);
  init_buffers<<<(NN * HIDDEN + 255) / 256, 256, 0, stream>>>(mbuf, denom, agg, NN);

  const int gblk = (NN + 127) / 128;
  const int eblk = (E * HEADS + 255) / 256;

  // fused QKV projection
  gemm_bf16_wmma<128, 384, false><<<gblk, 256, 0, stream>>>(x, wqkvF, bqkv, qkv, NN);
  // segment softmax attention
  edge_alpha<<<eblk, 256, 0, stream>>>(qkv, ei, alpha, mbuf, E);
  edge_exp<<<eblk, 256, 0, stream>>>(ei, alpha, mbuf, denom, E);
  edge_agg<<<eblk, 256, 0, stream>>>(qkv, ei, alpha, denom, agg, E);
  // output projection + residual LN1
  gemm_bf16_wmma<128, 128, false><<<gblk, 256, 0, stream>>>(agg, woF, bo, att, NN);
  add_ln<<<(NN + 7) / 8, 256, 0, stream>>>(x, att, g1, be1, hbuf, NN);
  // FFN with exact-GELU epilogue + residual LN2
  gemm_bf16_wmma<128, 512, true><<<gblk, 256, 0, stream>>>(hbuf, w1F, b1, tmid, NN);
  gemm_bf16_wmma<512, 128, false><<<gblk, 256, 0, stream>>>(tmid, w2F, b2, att, NN);
  add_ln<<<(NN + 7) / 8, 256, 0, stream>>>(hbuf, att, g2, be2, (float*)d_out, NN);
}
